// AttentionHead_39608188403928
// MI455X (gfx1250) — compile-verified, matrix operand order checked by reading, not run on hardware
//
#include <hip/hip_runtime.h>
#include <hip/hip_bf16.h>

// ---- problem constants (from reference) ----
#define NB 4
#define LL 2048
#define HH 1024
#define DD 64

// 0.125 (=1/sqrt(64)) * log2(e): folded into the Q projection so the
// softmax can run directly on hardware base-2 exp (v_exp_f32).
#define QSCALE 0.18033688011112042f

typedef __attribute__((ext_vector_type(16))) __bf16 v16bf;
typedef __attribute__((ext_vector_type(8)))  float  v8f;

__device__ __forceinline__ v8f vzero8() {
    v8f z;
#pragma unroll
    for (int i = 0; i < 8; ++i) z[i] = 0.0f;
    return z;
}

// A/B 16-bit fragment K-index pattern (wave32):
// lane<16 holds K {0..7,16..23}, lane>=16 holds K {8..15,24..31}
__device__ __forceinline__ int fragK(int e, int kb) {
    return kb + e + ((e & 8) ? 8 : 0);
}

// CDNA5 async global->LDS copy (16 bytes per lane), tracked by ASYNCcnt.
// Low 32 bits of a generic pointer into LDS == wave-relative LDS byte offset
// (ISA 10.2 aperture mapping), which is what VDST wants.
__device__ __forceinline__ void async_ld_b128(const void* lds_ptr, const void* gsrc) {
    unsigned loff = (unsigned)(unsigned long long)lds_ptr;
    asm volatile("global_load_async_to_lds_b128 %0, %1, off"
                 :: "v"(loff), "v"(gsrc)
                 : "memory");
}
__device__ __forceinline__ void wait_async0() {
    asm volatile("s_wait_asynccnt 0" ::: "memory");
}

// ============================================================
// Kernel 0: convert W (f32, [1024][64]) -> WT (bf16, [3][64][1024])
// grid = 768, block = 256   (3 * 1024 * 64 elements)
// ============================================================
__global__ __launch_bounds__(256) void wconv_kernel(
    const float* __restrict__ Wq, const float* __restrict__ Wk,
    const float* __restrict__ Wv, __bf16* __restrict__ WT)
{
    const int idx = blockIdx.x * 256 + threadIdx.x;   // 0 .. 3*65536-1
    const int mat = idx >> 16;                        // 65536 = 1024*64
    const int r   = idx & 65535;
    const int k   = r >> 6;
    const int n   = r & 63;
    const float* W = (mat == 0) ? Wq : (mat == 1) ? Wk : Wv;
    WT[(size_t)mat * 64 * HH + (size_t)n * HH + k] = (__bf16)W[(size_t)k * DD + n];
}

// ============================================================
// Kernel 1: fused QKV projection.  grid = (64, 3), block = 256
//   X(8192x1024) * W(1024x64) + b  ->  bf16
//   mat 0 -> qb [m][d] (scaled by QSCALE) ; mat 1 -> kbr [m][d] ;
//   mat 2 -> vbT [n][d][l]
// Double-buffered async LDS staging.
// ============================================================
__global__ __launch_bounds__(256) void qkv_proj_kernel(
    const float* __restrict__ q_in, const float* __restrict__ k_in,
    const float* __restrict__ v_in,
    const __bf16* __restrict__ WT,                 // [3][64][1024] bf16
    const float* __restrict__ bq, const float* __restrict__ bk,
    const float* __restrict__ bv,
    __bf16* __restrict__ qb, __bf16* __restrict__ kbr, __bf16* __restrict__ vbT)
{
    const int mat = blockIdx.y;                    // 0=q, 1=k, 2=v
    const float*  X   = (mat == 0) ? q_in : (mat == 1) ? k_in : v_in;
    const __bf16* WTm = WT + (size_t)mat * 64 * HH;
    const float*  Bb  = (mat == 0) ? bq : (mat == 1) ? bk : bv;

    const int mBase     = blockIdx.x * 128;        // row base in flattened (N*L)
    const int tid       = threadIdx.x;
    const int lane      = tid & 31;
    const int wave      = tid >> 5;                // 0..7
    const int rowInTile = lane & 15;
    const int kb        = (lane & 16) ? 8 : 0;
    const int hb        = (lane & 16) ? 8 : 0;     // C-layout row offset

    __shared__ __align__(16) float  sA[2][128 * 32];   // [row][k] f32
    __shared__ __align__(16) __bf16 sWT[2][64 * 32];   // [n][k]  bf16

    auto stage = [&](int kk, int bufi) {
#pragma unroll
        for (int it = 0; it < 4; ++it) {
            int idx = (it * 256 + tid) * 4;
            int r = idx >> 5, c = idx & 31;
            async_ld_b128(&sA[bufi][r * 32 + c],
                          X + (size_t)(mBase + r) * HH + kk + c);
        }
        {
            int idx = tid * 8;
            int nn = idx >> 5, c = idx & 31;
            async_ld_b128(&sWT[bufi][nn * 32 + c],
                          WTm + (size_t)nn * HH + kk + c);
        }
    };

    v8f acc[4];
#pragma unroll
    for (int nt = 0; nt < 4; ++nt) acc[nt] = vzero8();

    stage(0, 0);
    wait_async0();
    __syncthreads();

    for (int kk = 0; kk < HH; kk += 32) {
        const int bufi = (kk >> 5) & 1;
        if (kk + 32 < HH) stage(kk + 32, bufi ^ 1);   // prefetch next tile

        // A fragment: 16x32 (two contiguous runs -> ds_load_b128 + cvt)
        v16bf a;
        const float* arow = &sA[bufi][(wave * 16 + rowInTile) * 32];
#pragma unroll
        for (int e = 0; e < 16; ++e) a[e] = (__bf16)arow[fragK(e, kb)];

        // 4 B fragments (bf16, no conversion) + WMMA
#pragma unroll
        for (int nt = 0; nt < 4; ++nt) {
            v16bf b;
            const __bf16* wrow = &sWT[bufi][(nt * 16 + rowInTile) * 32];
#pragma unroll
            for (int e = 0; e < 16; ++e) b[e] = wrow[fragK(e, kb)];
            acc[nt] = __builtin_amdgcn_wmma_f32_16x16x32_bf16(
                false, a, false, b, (short)0, acc[nt], false, false);
        }
        wait_async0();        // next tile landed
        __syncthreads();      // everyone done reading current tile
    }

    // epilogue: bias (+ softmax scale folded into Q) + bf16 store
    const int mTop = mBase + wave * 16;
#pragma unroll
    for (int nt = 0; nt < 4; ++nt) {
        const int d = nt * 16 + rowInTile;         // C layout: col = lane%16
        const float bias = Bb[d];
#pragma unroll
        for (int i = 0; i < 8; ++i) {
            const int m = mTop + hb + i;           // C layout: row = i (+8 hi half)
            float val = acc[nt][i] + bias;
            if (mat == 0) val *= QSCALE;           // fold 0.125*log2(e) into q
            const __bf16 h = (__bf16)val;
            const int n = m >> 11;                 // /L (2048)
            const int l = m & (LL - 1);
            if (mat == 2) {
                vbT[((size_t)n * DD + d) * LL + l] = h;
            } else {
                __bf16* dst = (mat == 0) ? qb : kbr;
                dst[(size_t)m * DD + d] = h;
            }
        }
    }
}

// ============================================================
// Kernel 2: flash attention.  grid = 64 (= N * L/128), block = 256
// Each wave: 16 query rows vs all 2048 keys, 32 keys per step.
// LDS: sK [key][d] (32x64), sVT [d][key] (64x32), sP per-wave [16][32]
// Double-buffered async LDS staging; softmax in exp2 domain.
// ============================================================
__global__ __launch_bounds__(256) void attn_kernel(
    const __bf16* __restrict__ qb, const __bf16* __restrict__ kbr,
    const __bf16* __restrict__ vbT, float* __restrict__ out)
{
    const int tid       = threadIdx.x;
    const int lane      = tid & 31;
    const int wave      = tid >> 5;
    const int n         = blockIdx.x >> 4;         // batch
    const int qt        = blockIdx.x & 15;
    const int qBase     = qt * 128;
    const int rowInTile = lane & 15;
    const int kb        = (lane & 16) ? 8 : 0;
    const int hb        = (lane & 16) ? 8 : 0;

    __shared__ __align__(16) __bf16 sK[2][32 * 64];   // [key][d]
    __shared__ __align__(16) __bf16 sVT[2][64 * 32];  // [d][key]
    __shared__ __align__(16) __bf16 sP[8][16 * 32];

    auto stageKV = [&](int kt, int bufi) {
        {
            int idx = tid * 8;
            int key = idx >> 6, c = idx & 63;
            async_ld_b128(&sK[bufi][key * 64 + c],
                          kbr + (size_t)(n * LL + kt + key) * DD + c);
        }
        {
            int idx = tid * 8;
            int d = idx >> 5, c = idx & 31;
            async_ld_b128(&sVT[bufi][d * 32 + c],
                          vbT + ((size_t)n * DD + d) * LL + kt + c);
        }
    };

    // load this wave's Q fragments (16 rows x 64 d -> two 16x32 A-frags)
    const __bf16* qrow =
        qb + (size_t)(n * LL + qBase + wave * 16 + rowInTile) * DD;
    v16bf qf[2];
#pragma unroll
    for (int h = 0; h < 2; ++h)
#pragma unroll
        for (int e = 0; e < 16; ++e)
            qf[h][e] = qrow[h * 32 + fragK(e, kb)];

    v8f oacc[4];
#pragma unroll
    for (int dt = 0; dt < 4; ++dt) oacc[dt] = vzero8();
    float mrow[8], lrow[8];
#pragma unroll
    for (int i = 0; i < 8; ++i) { mrow[i] = -1e30f; lrow[i] = 0.0f; }

    stageKV(0, 0);
    wait_async0();
    __syncthreads();

    for (int kt = 0; kt < LL; kt += 32) {
        const int bufi = (kt >> 5) & 1;
        if (kt + 32 < LL) stageKV(kt + 32, bufi ^ 1);   // prefetch next tile
        // deepen the pipeline: warm L2 for the tile after next
        if (kt + 64 < LL) {
            __builtin_prefetch(kbr + (size_t)(n * LL + kt + 64 + (tid & 31)) * DD, 0, 0);
        }

        // scores S = Q K^T (pre-scaled by 0.125*log2e): 2 key tiles x 2 d-steps
        v8f sc[2];
        sc[0] = vzero8(); sc[1] = vzero8();
#pragma unroll
        for (int ct = 0; ct < 2; ++ct) {
            const int keyc = ct * 16 + rowInTile;
#pragma unroll
            for (int h = 0; h < 2; ++h) {
                v16bf bkf;
#pragma unroll
                for (int e = 0; e < 16; ++e)
                    bkf[e] = sK[bufi][keyc * 64 + h * 32 + fragK(e, kb)];
                sc[ct] = __builtin_amdgcn_wmma_f32_16x16x32_bf16(
                    false, qf[h], false, bkf, (short)0, sc[ct], false, false);
            }
        }

        // online softmax in exp2 domain; C layout: vgpr i -> row (hb+i)
        float corr[8];
#pragma unroll
        for (int i = 0; i < 8; ++i) {
            const float s0 = sc[0][i], s1 = sc[1][i];
            float tmax = fmaxf(s0, s1);
#pragma unroll
            for (int off = 1; off < 16; off <<= 1)
                tmax = fmaxf(tmax, __shfl_xor(tmax, off, 32));
            const float mnew = fmaxf(mrow[i], tmax);
            const float p0 = exp2f(s0 - mnew);     // v_exp_f32 is base-2
            const float p1 = exp2f(s1 - mnew);
            float psum = p0 + p1;
#pragma unroll
            for (int off = 1; off < 16; off <<= 1)
                psum += __shfl_xor(psum, off, 32);
            const float c = exp2f(mrow[i] - mnew);
            corr[i] = c;
            lrow[i] = lrow[i] * c + psum;
            mrow[i] = mnew;
            // spill P (C layout) to per-wave LDS tile [q][key]
            sP[wave][(hb + i) * 32 + rowInTile]      = (__bf16)p0;
            sP[wave][(hb + i) * 32 + 16 + rowInTile] = (__bf16)p1;
        }
#pragma unroll
        for (int dt = 0; dt < 4; ++dt)
#pragma unroll
            for (int i = 0; i < 8; ++i) oacc[dt][i] *= corr[i];

        // reload P in A-fragment layout (same-wave DS ordering; no barrier)
        v16bf pf;
#pragma unroll
        for (int e = 0; e < 16; ++e)
            pf[e] = sP[wave][rowInTile * 32 + fragK(e, kb)];

        // O += P V : 4 d-tiles (contiguous key runs -> ds_load_b128)
#pragma unroll
        for (int dt = 0; dt < 4; ++dt) {
            const int d = dt * 16 + rowInTile;
            v16bf bvf;
#pragma unroll
            for (int e = 0; e < 16; ++e)
                bvf[e] = sVT[bufi][d * 32 + fragK(e, kb)];
            oacc[dt] = __builtin_amdgcn_wmma_f32_16x16x32_bf16(
                false, pf, false, bvf, (short)0, oacc[dt], false, false);
        }
        wait_async0();        // next K/V tile landed
        __syncthreads();      // everyone done reading current tile
    }

    // epilogue: normalize and store f32 output (N, L, 64)
    float* orow = out + (size_t)(n * LL + qBase + wave * 16) * DD;
#pragma unroll
    for (int dt = 0; dt < 4; ++dt) {
        const int d = dt * 16 + rowInTile;
#pragma unroll
        for (int i = 0; i < 8; ++i) {
            const float inv = 1.0f / lrow[i];
            orow[(size_t)(hb + i) * DD + d] = oacc[dt][i] * inv;
        }
    }
}

// ============================================================
extern "C" void kernel_launch(void* const* d_in, const int* in_sizes, int n_in,
                              void* d_out, int out_size, void* d_ws, size_t ws_size,
                              hipStream_t stream) {
    const float* query = (const float*)d_in[0];
    const float* key_  = (const float*)d_in[1];
    const float* value = (const float*)d_in[2];
    const float* Wq    = (const float*)d_in[3];
    const float* bq    = (const float*)d_in[4];
    const float* Wk    = (const float*)d_in[5];
    const float* bk    = (const float*)d_in[6];
    const float* Wv    = (const float*)d_in[7];
    const float* bv    = (const float*)d_in[8];
    float* out = (float*)d_out;

    const size_t elems = (size_t)NB * LL * DD;     // 524288
    __bf16* qb  = (__bf16*)d_ws;
    __bf16* kbr = qb + elems;
    __bf16* vbT = kbr + elems;
    __bf16* WT  = vbT + elems;                     // [3][64][1024]

    wconv_kernel<<<768, 256, 0, stream>>>(Wq, Wk, Wv, WT);
    dim3 g1(64, 3, 1);
    qkv_proj_kernel<<<g1, 256, 0, stream>>>(query, key_, value, WT,
                                            bq, bk, bv, qb, kbr, vbT);
    attn_kernel<<<64, 256, 0, stream>>>(qb, kbr, vbT, out);
}
